// TransformerEncoderLayerMoE_86517821216095
// MI455X (gfx1250) — matrix operand order repaired and checked by reading, not compiled, not run on hardware
//
#include <hip/hip_runtime.h>
#include <hip/hip_bf16.h>

// ---------------------------------------------------------------------------
// TransformerEncoderLayer + MoE for MI455X (gfx1250, wave32, WMMA bf16)
// S=1024 B=4 D=1024 H=16 dh=64 E=8 K=2 F=4096, N = S*B = 4096 tokens
//   1) fp32 -> bf16 converts
//   2) QKV GEMM (WMMA, async double-buffered LDS)  -> qkv_bf [N,3D]
//   3) flash attention (WMMA)                      -> ctx_bf [N,D]
//   4) out-proj GEMM                               -> attn_f32 [N,D]
//   5) residual + LN1                              -> x_f32, x_bf
//   6) gating: softmax/top2 + atomic expert bucketing
//   7) per expert: gather-GEMM(relu) -> h_bf ; GEMM + scaled scatter-atomicAdd
//   8) residual + LN2 -> d_out (f32)
// GEMM global->LDS uses GLOBAL_LOAD_ASYNC_TO_LDS_B128 + s_wait_asynccnt
// (CDNA5 async path) with 2-deep LDS double buffering.
// ---------------------------------------------------------------------------

#define S_LEN 1024
#define BSZ   4
#define DMODEL 1024
#define NHEAD 16
#define DHEAD 64
#define NEXP  8
#define FDIM  4096
#define NTOK  4096          // S*B
#define TDIM  3072          // 3*D

typedef __attribute__((ext_vector_type(16))) __bf16 v16bf;
typedef __attribute__((ext_vector_type(8)))  float  v8f;

struct BF32 { uint4 a, b; };   // 32 bytes = 16 bf16

static __device__ __forceinline__ __bf16 f2bf(float f) {
    unsigned u = __builtin_bit_cast(unsigned, f);
    unsigned r = u + 0x7FFFu + ((u >> 16) & 1u);       // round-to-nearest-even
    unsigned short h = (unsigned short)(r >> 16);
    return __builtin_bit_cast(__bf16, h);
}

// 16 contiguous bf16 (32B)
static __device__ __forceinline__ v16bf ld32(const __bf16* p) {
    BF32 t; t.a = *(const uint4*)p; t.b = *(const uint4*)(p + 8);
    return __builtin_bit_cast(v16bf, t);
}
// two 16B chunks at p and p+16 elems (A-fragment lane pattern)
static __device__ __forceinline__ v16bf ld2x16(const __bf16* p) {
    BF32 t; t.a = *(const uint4*)p; t.b = *(const uint4*)(p + 16);
    return __builtin_bit_cast(v16bf, t);
}

static __device__ __forceinline__ v8f wmma_bf16(v16bf a, v16bf b, v8f c) {
    return __builtin_amdgcn_wmma_f32_16x16x32_bf16(false, a, false, b, (short)0, c,
                                                   false, false);
}

// CDNA5 async global->LDS copy, 16B per lane, tracked by ASYNCcnt.
// lds_off = LDS byte address (low 32 bits of flat shared pointer).
static __device__ __forceinline__ void async_ld16(unsigned lds_off,
                                                  const __bf16* gptr) {
    asm volatile("global_load_async_to_lds_b128 %0, %1, off"
                 :: "v"(lds_off), "v"(gptr) : "memory");
}
static __device__ __forceinline__ unsigned lds_addr_of(const void* p) {
    return (unsigned)(unsigned long long)p;   // flat LDS aperture: offset in [31:0]
}

static __device__ __forceinline__ float redmax16(float v) {
#pragma unroll
    for (int off = 1; off < 16; off <<= 1) v = fmaxf(v, __shfl_xor(v, off, 32));
    return v;
}
static __device__ __forceinline__ float redsum16(float v) {
#pragma unroll
    for (int off = 1; off < 16; off <<= 1) v += __shfl_xor(v, off, 32);
    return v;
}

// ---------------------------------------------------------------------------
// fp32 -> bf16 conversion
// ---------------------------------------------------------------------------
__global__ __launch_bounds__(256) void cvt_bf16_kernel(const float* __restrict__ in,
                                                       __bf16* __restrict__ out,
                                                       long long n) {
    long long i = (long long)blockIdx.x * blockDim.x + threadIdx.x;
    long long stride = (long long)gridDim.x * blockDim.x;
    for (; i < n; i += stride) out[i] = f2bf(in[i]);
}

// ---------------------------------------------------------------------------
// Tiled bf16 WMMA GEMM:  out[M,Nc] = A[M,Kd] @ Bw[Nc,Kd]^T + bias
// Block: 256 thr = 8 waves, 128x128 tile; wave = 64x32 (4x2 WMMA tiles).
// Global->LDS via async-to-LDS, 2-deep double buffer:
//   prefetch(next); s_wait_asynccnt 4  -> current buffer landed
//   barrier; 8 WMMAs from current; barrier (so next iter may overwrite it)
// ---------------------------------------------------------------------------
#define GBM 128
#define GBN 128
#define GKT 32
#define GLD 40   // padded LDS row stride (bf16 elems)

template <bool RELU, bool GATHER, bool SCATTER>
__global__ __launch_bounds__(256)
void gemm_bf16_wmma(const __bf16* __restrict__ A, const __bf16* __restrict__ Bw,
                    const float* __restrict__ bias, int M, int Nc, int Kd,
                    const int* __restrict__ cntPtr, const int* __restrict__ rowmap,
                    const float* __restrict__ rowscale,
                    __bf16* __restrict__ outb, float* __restrict__ outf, int ldo) {
    __shared__ __bf16 lds_a[2][GBM * GLD];
    __shared__ __bf16 lds_b[2][GBN * GLD];

    const int Mreal = cntPtr ? *cntPtr : M;
    const int m_base = blockIdx.y * GBM;
    if (m_base >= Mreal) return;
    const int n_base = blockIdx.x * GBN;

    const int tid  = threadIdx.x;
    const int lane = tid & 31;
    const int wid  = tid >> 5;
    const int wm   = (wid & 1) * 64;    // wave row offset in tile
    const int wn   = (wid >> 1) * 32;   // wave col offset in tile
    const int lm   = lane & 15;
    const int lhi  = lane >> 4;

    // cooperative loader coords: each thread owns one 32B chunk per matrix
    const int crow  = tid >> 1;         // 0..127
    const int chalf = tid & 1;          // 0 or 1 (16 bf16 each)
    int arow_t = m_base + crow;
    if (arow_t > Mreal - 1) arow_t = Mreal - 1;
    const long long arow = GATHER ? (long long)rowmap[arow_t] : (long long)arow_t;
    const long long brow = (long long)(n_base + crow);

    const __bf16* agp = A + arow * Kd + chalf * 16;
    const __bf16* bgp = Bw + brow * Kd + chalf * 16;
    const unsigned la[2] = { lds_addr_of(&lds_a[0][crow * GLD + chalf * 16]),
                             lds_addr_of(&lds_a[1][crow * GLD + chalf * 16]) };
    const unsigned lb[2] = { lds_addr_of(&lds_b[0][crow * GLD + chalf * 16]),
                             lds_addr_of(&lds_b[1][crow * GLD + chalf * 16]) };

    v8f acc[4][2] = {};

    // prologue prefetch of K-tile 0 into buffer 0 (4 asyncs per wave)
    async_ld16(la[0],      agp);
    async_ld16(la[0] + 16, agp + 8);
    async_ld16(lb[0],      bgp);
    async_ld16(lb[0] + 16, bgp + 8);

    const int nk = Kd / GKT;
    for (int i = 0; i < nk; ++i) {
        const int cur = i & 1;
        if (i + 1 < nk) {
            const int nxt = cur ^ 1;
            const int k1 = (i + 1) * GKT;
            async_ld16(la[nxt],      agp + k1);
            async_ld16(la[nxt] + 16, agp + k1 + 8);
            async_ld16(lb[nxt],      bgp + k1);
            async_ld16(lb[nxt] + 16, bgp + k1 + 8);
            // only the 4 just-issued asyncs may remain outstanding
            asm volatile("s_wait_asynccnt 0x4" ::: "memory");
        } else {
            asm volatile("s_wait_asynccnt 0x0" ::: "memory");
        }
        __syncthreads();   // current buffer visible to all waves

        const __bf16* ta = lds_a[cur];
        const __bf16* tb = lds_b[cur];
        v16bf af[4], bf[2];
#pragma unroll
        for (int mt = 0; mt < 4; ++mt)
            af[mt] = ld2x16(ta + (wm + mt * 16 + lm) * GLD + lhi * 8);
#pragma unroll
        for (int nt = 0; nt < 2; ++nt)
            bf[nt] = ld2x16(tb + (wn + nt * 16 + lm) * GLD + lhi * 8);
#pragma unroll
        for (int mt = 0; mt < 4; ++mt)
#pragma unroll
            for (int nt = 0; nt < 2; ++nt)
                acc[mt][nt] = wmma_bf16(af[mt], bf[nt], acc[mt][nt]);

        __syncthreads();   // everyone done reading before it is overwritten
    }

    // epilogue
#pragma unroll
    for (int nt = 0; nt < 2; ++nt) {
        const int gcol = n_base + wn + nt * 16 + lm;
        const float bv = bias ? bias[gcol] : 0.f;
#pragma unroll
        for (int mt = 0; mt < 4; ++mt) {
#pragma unroll
            for (int r = 0; r < 8; ++r) {
                const int trow = m_base + wm + mt * 16 + lhi * 8 + r;
                float v = acc[mt][nt][r] + bv;
                if (RELU) v = fmaxf(v, 0.f);
                if (trow < Mreal) {
                    if (SCATTER) {
                        const long long orow = (long long)rowmap[trow];
                        atomicAdd(outf + orow * ldo + gcol, rowscale[trow] * v);
                    } else {
                        if (outf) outf[(long long)trow * ldo + gcol] = v;
                        if (outb) outb[(long long)trow * ldo + gcol] = f2bf(v);
                    }
                }
            }
        }
    }
}

// ---------------------------------------------------------------------------
// Flash attention: grid (S/64, B*H), block 128 (4 waves). Each wave owns a
// 16-row Q tile, block shares V^T staging. dh = 64, keys swept 32 at a time.
// ---------------------------------------------------------------------------
__global__ __launch_bounds__(128)
void attn_flash_kernel(const __bf16* __restrict__ qkv, __bf16* __restrict__ ctx) {
    __shared__ __bf16 ldsVT[64 * GLD];        // [dh][key] for current 32 keys
    __shared__ __bf16 ldsP[4][16 * GLD];      // per-wave P tile [16 q][32 k]

    const int bh = blockIdx.y;
    const int h = bh & (NHEAD - 1);
    const int b = bh >> 4;
    const int wid  = threadIdx.x >> 5;
    const int lane = threadIdx.x & 31;
    const int lm   = lane & 15;
    const int lhi  = lane >> 4;
    const int qtile = blockIdx.x * 64 + wid * 16;

    // Q fragments (16x32 per dh chunk), loaded straight from global
    const long long qrow = (long long)(qtile + lm) * BSZ + b;
    const __bf16* qp = qkv + qrow * TDIM + h * DHEAD;
    v16bf qa[2];
    qa[0] = ld2x16(qp + lhi * 8);
    qa[1] = ld2x16(qp + 32 + lhi * 8);

    v8f O[4] = {};
    float mrow[8], lrow[8];
#pragma unroll
    for (int r = 0; r < 8; ++r) { mrow[r] = -1e30f; lrow[r] = 0.f; }

    for (int j0 = 0; j0 < S_LEN; j0 += 32) {
        __syncthreads();   // protect LDS reuse from previous iteration
        {   // stage V^T [64 dh][32 keys] cooperatively (128 threads)
            const int key = threadIdx.x >> 2;   // 0..31
            const int ch  = threadIdx.x & 3;    // dh chunk of 16
            const long long krow = (long long)(j0 + key) * BSZ + b;
            const __bf16* vp = qkv + krow * TDIM + 2 * DMODEL + h * DHEAD + ch * 16;
            __bf16 tmp[16];
            *(uint4*)&tmp[0] = *(const uint4*)vp;
            *(uint4*)&tmp[8] = *(const uint4*)(vp + 8);
#pragma unroll
            for (int i = 0; i < 16; ++i)
                ldsVT[(ch * 16 + i) * GLD + key] = tmp[i];
        }

        // scores S[16q x 32k] = Q K^T : B-fragments read straight from global
        v8f S0 = {}, S1 = {};
#pragma unroll
        for (int c = 0; c < 2; ++c) {
            const long long k0r = (long long)(j0 + lm) * BSZ + b;
            const long long k1r = (long long)(j0 + 16 + lm) * BSZ + b;
            v16bf kb0 = ld2x16(qkv + k0r * TDIM + DMODEL + h * DHEAD + c * 32 + lhi * 8);
            v16bf kb1 = ld2x16(qkv + k1r * TDIM + DMODEL + h * DHEAD + c * 32 + lhi * 8);
            S0 = wmma_bf16(qa[c], kb0, S0);
            S1 = wmma_bf16(qa[c], kb1, S1);
        }

        // online softmax update (row r lives on this lane's 16-lane half)
        float fac[8];
#pragma unroll
        for (int r = 0; r < 8; ++r) {
            float a0 = S0[r] * 0.125f;     // dh^-0.5 = 1/8
            float a1 = S1[r] * 0.125f;
            float mx = redmax16(fmaxf(a0, a1));
            float nm = fmaxf(mrow[r], mx);
            fac[r] = __expf(mrow[r] - nm);
            float p0 = __expf(a0 - nm);
            float p1 = __expf(a1 - nm);
            S0[r] = p0; S1[r] = p1;
            lrow[r] = lrow[r] * fac[r] + redsum16(p0 + p1);
            mrow[r] = nm;
        }
#pragma unroll
        for (int d = 0; d < 4; ++d)
#pragma unroll
            for (int r = 0; r < 8; ++r) O[d][r] *= fac[r];

        // re-layout P (C-frag) -> A-frag via LDS
        __bf16* pl = ldsP[wid];
#pragma unroll
        for (int r = 0; r < 8; ++r) {
            const int row = lhi * 8 + r;
            pl[row * GLD + lm]      = f2bf(S0[r]);
            pl[row * GLD + 16 + lm] = f2bf(S1[r]);
        }
        __syncthreads();   // V^T staged + P written

        v16bf pa = ld2x16(pl + lm * GLD + lhi * 8);
#pragma unroll
        for (int d = 0; d < 4; ++d) {
            v16bf vb = ld2x16(ldsVT + (d * 16 + lm) * GLD + lhi * 8);
            O[d] = wmma_bf16(pa, vb, O[d]);
        }
    }

    // normalize + store ctx (bf16)
#pragma unroll
    for (int d = 0; d < 4; ++d) {
#pragma unroll
        for (int r = 0; r < 8; ++r) {
            const int row = lhi * 8 + r;
            const long long token = (long long)(qtile + row) * BSZ + b;
            ctx[token * DMODEL + h * DHEAD + d * 16 + lm] = f2bf(O[d][r] / lrow[r]);
        }
    }
}

// ---------------------------------------------------------------------------
// residual + LayerNorm : out = LN(a + b) * g + beta   (one block per row)
// ---------------------------------------------------------------------------
__global__ __launch_bounds__(256)
void residual_ln_kernel(const float* __restrict__ a, const float* __restrict__ b,
                        const float* __restrict__ g, const float* __restrict__ be,
                        float* __restrict__ outf, __bf16* __restrict__ outbf, int Dd) {
    __shared__ float s1[256], s2[256], stats[2];
    const long long row = blockIdx.x;
    const float* ar = a + row * Dd;
    const float* br = b + row * Dd;
    float ls = 0.f, lq = 0.f;
    for (int d = threadIdx.x; d < Dd; d += 256) {
        float v = ar[d] + br[d];
        ls += v; lq += v * v;
    }
    s1[threadIdx.x] = ls; s2[threadIdx.x] = lq;
    __syncthreads();
    for (int off = 128; off > 0; off >>= 1) {
        if (threadIdx.x < off) {
            s1[threadIdx.x] += s1[threadIdx.x + off];
            s2[threadIdx.x] += s2[threadIdx.x + off];
        }
        __syncthreads();
    }
    if (threadIdx.x == 0) {
        float mean = s1[0] / Dd;
        float var  = s2[0] / Dd - mean * mean;
        stats[0] = mean;
        stats[1] = rsqrtf(var + 1e-5f);
    }
    __syncthreads();
    const float mean = stats[0], rstd = stats[1];
    for (int d = threadIdx.x; d < Dd; d += 256) {
        float v = (ar[d] + br[d] - mean) * rstd * g[d] + be[d];
        if (outf)  outf[row * Dd + d]  = v;
        if (outbf) outbf[row * Dd + d] = f2bf(v);
    }
}

// ---------------------------------------------------------------------------
// Gating: logits -> softmax -> top2 -> normalized weights; atomic bucketing
// ---------------------------------------------------------------------------
__global__ __launch_bounds__(256)
void gate_topk_kernel(const float* __restrict__ x, const float* __restrict__ gw,
                      const float* __restrict__ gb, int* __restrict__ cnt,
                      int* __restrict__ toks, float* __restrict__ scl) {
    const int t = blockIdx.x * blockDim.x + threadIdx.x;
    if (t >= NTOK) return;
    float acc[NEXP];
#pragma unroll
    for (int e = 0; e < NEXP; ++e) acc[e] = gb[e];
    const float* xr = x + (long long)t * DMODEL;
    for (int d = 0; d < DMODEL; ++d) {
        float xv = xr[d];
#pragma unroll
        for (int e = 0; e < NEXP; ++e) acc[e] += xv * gw[e * DMODEL + d];
    }
    float mx = acc[0];
#pragma unroll
    for (int e = 1; e < NEXP; ++e) mx = fmaxf(mx, acc[e]);
    float p[NEXP];
#pragma unroll
    for (int e = 0; e < NEXP; ++e) p[e] = __expf(acc[e] - mx);
    int i1 = 0;
#pragma unroll
    for (int e = 1; e < NEXP; ++e) if (p[e] > p[i1]) i1 = e;
    int i2 = (i1 == 0) ? 1 : 0;
#pragma unroll
    for (int e = 0; e < NEXP; ++e) if (e != i2 && e != i1 && p[e] > p[i2]) i2 = e;
    const float denom = p[i1] + p[i2];
    const float w1 = p[i1] / denom;
    const float w2 = p[i2] / denom;
    int s1 = atomicAdd(&cnt[i1], 1);
    toks[i1 * NTOK + s1] = t; scl[i1 * NTOK + s1] = w1;
    int s2 = atomicAdd(&cnt[i2], 1);
    toks[i2 * NTOK + s2] = t; scl[i2 * NTOK + s2] = w2;
}

// ---------------------------------------------------------------------------
// Host orchestration
// ---------------------------------------------------------------------------
extern "C" void kernel_launch(void* const* d_in, const int* in_sizes, int n_in,
                              void* d_out, int out_size, void* d_ws, size_t ws_size,
                              hipStream_t stream) {
    (void)in_sizes; (void)n_in; (void)out_size; (void)ws_size;
    const float* src       = (const float*)d_in[0];
    const float* in_proj_w = (const float*)d_in[1];
    const float* in_proj_b = (const float*)d_in[2];
    const float* out_w     = (const float*)d_in[3];
    const float* out_b     = (const float*)d_in[4];
    const float* gate_w    = (const float*)d_in[5];
    const float* gate_b    = (const float*)d_in[6];
    const float* w1        = (const float*)d_in[7];
    const float* b1        = (const float*)d_in[8];
    const float* w2        = (const float*)d_in[9];
    const float* b2        = (const float*)d_in[10];
    const float* ln1_g     = (const float*)d_in[11];
    const float* ln1_b     = (const float*)d_in[12];
    const float* ln2_g     = (const float*)d_in[13];
    const float* ln2_b     = (const float*)d_in[14];

    char* wp = (char*)d_ws;
    auto alloc = [&](size_t bytes) {
        void* p = (void*)wp;
        wp += (bytes + 255) & ~(size_t)255;
        return p;
    };
    __bf16* src_bf  = (__bf16*)alloc((size_t)NTOK * DMODEL * 2);
    __bf16* wqkv_bf = (__bf16*)alloc((size_t)TDIM * DMODEL * 2);
    __bf16* wout_bf = (__bf16*)alloc((size_t)DMODEL * DMODEL * 2);
    __bf16* w1_bf   = (__bf16*)alloc((size_t)NEXP * FDIM * DMODEL * 2);
    __bf16* w2_bf   = (__bf16*)alloc((size_t)NEXP * DMODEL * FDIM * 2);
    __bf16* qkv_bf  = (__bf16*)alloc((size_t)NTOK * TDIM * 2);
    __bf16* ctx_bf  = (__bf16*)alloc((size_t)NTOK * DMODEL * 2);
    float*  attn_f  = (float*)alloc((size_t)NTOK * DMODEL * 4);
    float*  x_f     = (float*)alloc((size_t)NTOK * DMODEL * 4);
    __bf16* x_bf    = (__bf16*)alloc((size_t)NTOK * DMODEL * 2);
    float*  moe_f   = (float*)alloc((size_t)NTOK * DMODEL * 4);
    __bf16* h_bf    = (__bf16*)alloc((size_t)NTOK * FDIM * 2);
    int*    cnt     = (int*)alloc((size_t)NEXP * 4);
    int*    toks    = (int*)alloc((size_t)NEXP * NTOK * 4);
    float*  scl     = (float*)alloc((size_t)NEXP * NTOK * 4);

    // 1) fp32 -> bf16 converts
    cvt_bf16_kernel<<<2048, 256, 0, stream>>>(src, src_bf, (long long)NTOK * DMODEL);
    cvt_bf16_kernel<<<2048, 256, 0, stream>>>(in_proj_w, wqkv_bf, (long long)TDIM * DMODEL);
    cvt_bf16_kernel<<<2048, 256, 0, stream>>>(out_w, wout_bf, (long long)DMODEL * DMODEL);
    cvt_bf16_kernel<<<4096, 256, 0, stream>>>(w1, w1_bf, (long long)NEXP * FDIM * DMODEL);
    cvt_bf16_kernel<<<4096, 256, 0, stream>>>(w2, w2_bf, (long long)NEXP * DMODEL * FDIM);

    // 2) QKV projection: qkv = src @ in_proj_w^T + b
    gemm_bf16_wmma<false, false, false>
        <<<dim3(TDIM / GBN, NTOK / GBM), 256, 0, stream>>>(
            src_bf, wqkv_bf, in_proj_b, NTOK, TDIM, DMODEL,
            nullptr, nullptr, nullptr, qkv_bf, nullptr, TDIM);

    // 3) attention
    attn_flash_kernel<<<dim3(S_LEN / 64, BSZ * NHEAD), 128, 0, stream>>>(qkv_bf, ctx_bf);

    // 4) out projection
    gemm_bf16_wmma<false, false, false>
        <<<dim3(DMODEL / GBN, NTOK / GBM), 256, 0, stream>>>(
            ctx_bf, wout_bf, out_b, NTOK, DMODEL, DMODEL,
            nullptr, nullptr, nullptr, nullptr, attn_f, DMODEL);

    // 5) residual + LN1
    residual_ln_kernel<<<NTOK, 256, 0, stream>>>(src, attn_f, ln1_g, ln1_b,
                                                 x_f, x_bf, DMODEL);

    // 6) gating + routing
    hipMemsetAsync(moe_f, 0, (size_t)NTOK * DMODEL * 4, stream);
    hipMemsetAsync(cnt, 0, (size_t)NEXP * 4, stream);
    gate_topk_kernel<<<NTOK / 256, 256, 0, stream>>>(x_f, gate_w, gate_b, cnt, toks, scl);

    // 7) experts (top-2 routed, grouped GEMMs)
    for (int e = 0; e < NEXP; ++e) {
        gemm_bf16_wmma<true, true, false>
            <<<dim3(FDIM / GBN, NTOK / GBM), 256, 0, stream>>>(
                x_bf, w1_bf + (size_t)e * FDIM * DMODEL, b1 + (size_t)e * FDIM,
                NTOK, FDIM, DMODEL, cnt + e, toks + (size_t)e * NTOK, nullptr,
                h_bf, nullptr, FDIM);
        gemm_bf16_wmma<false, false, true>
            <<<dim3(DMODEL / GBN, NTOK / GBM), 256, 0, stream>>>(
                h_bf, w2_bf + (size_t)e * DMODEL * FDIM, b2 + (size_t)e * DMODEL,
                NTOK, DMODEL, FDIM, cnt + e, toks + (size_t)e * NTOK,
                scl + (size_t)e * NTOK, nullptr, moe_f, DMODEL);
    }

    // 8) residual + LN2 -> output
    residual_ln_kernel<<<NTOK, 256, 0, stream>>>(x_f, moe_f, ln2_g, ln2_b,
                                                 (float*)d_out, nullptr, DMODEL);
}
